// GroupedConv2d_57208964383022
// MI455X (gfx1250) — matrix operand order repaired and checked by reading, not compiled, not run on hardware
//
#include <hip/hip_runtime.h>

// ---- problem constants (from reference) ----
#define G_     8
#define C_PER  32
#define C_IN   256
#define J_     64
#define B_     16
#define H_     56
#define W_     56
#define KK     3
#define KDIM   (C_PER * KK * KK)     // 288
#define NPIX   (B_ * H_ * W_)        // 50176
#define NT     64                    // N pixels per workgroup (50176 / 64 = 784, exact)
#define LDK    292                   // padded K stride (elems); 584B row -> 2-way conflicts, 8B aligned
#define KSTEPS (KDIM / 32)           // 9 bf16-WMMA K steps

typedef __bf16  bf16x16 __attribute__((ext_vector_type(16)));
typedef float   f32x8   __attribute__((ext_vector_type(8)));

// async-to-LDS builtin pointer types (from hipcc diagnostic: AS1 "int __vector(4) __device__ *")
typedef int v4i_ __attribute__((vector_size(16)));
typedef __attribute__((address_space(1))) v4i_ g_v4i;   // global
typedef __attribute__((address_space(3))) v4i_ l_v4i;   // LDS

#if defined(__gfx1250__) && __has_builtin(__builtin_amdgcn_global_load_async_to_lds_b128)
#define HAVE_ASYNC_LDS 1
#else
#define HAVE_ASYNC_LDS 0
#endif

union FragU {
    unsigned int u[8];
    bf16x16      v;
};

static __device__ __forceinline__ unsigned short f2bf(float f) {
    unsigned int u = __builtin_bit_cast(unsigned int, f);
    unsigned int r = u + 0x7FFFu + ((u >> 16) & 1u);   // round-to-nearest-even
    return (unsigned short)(r >> 16);
}

static __device__ __forceinline__ void wait_async0() {
#if __has_builtin(__builtin_amdgcn_s_wait_asynccnt)
    __builtin_amdgcn_s_wait_asynccnt(0);
#else
    asm volatile("s_wait_asynccnt 0x0" ::: "memory");
#endif
}

// ---- prepass: whole weight tensor f32 -> bf16 (runs once per launch, 288KB) ----
__global__ __launch_bounds__(256) void convert_weights_bf16(
    const float* __restrict__ wgt, unsigned short* __restrict__ wbf, int n)
{
    int i = blockIdx.x * 256 + threadIdx.x;
    if (i < n) wbf[i] = f2bf(wgt[i]);
}

__global__ __launch_bounds__(256) void grouped_conv_wmma_bf16(
    const float*          __restrict__ x,      // [B, C_IN, H, W]
    const float*          __restrict__ wgt,    // [G, J, C_PER, 3, 3] (fallback path)
    const float*          __restrict__ bias,   // [G, J]
    const int*            __restrict__ arr,    // [G, C_PER]
    const unsigned short* __restrict__ wbf,    // pre-converted bf16 weights (may be unused)
    int                   use_pre,
    float*                __restrict__ out)    // [B, G*J, H, W]
{
    extern __shared__ char smem[];
    unsigned short* sA = (unsigned short*)smem;              // [64][LDK]  weights (bf16)
    unsigned short* sB = sA + J_ * LDK;                      // [64][LDK]  im2col  (bf16)

    const int tid  = threadIdx.x;
    const int lane = tid & 31;
    const int wid  = tid >> 5;       // 0..7
    const int g    = blockIdx.y;     // 0..7
    const int n0   = blockIdx.x * NT;

    // ---------------- Phase 1a: weights -> LDS ----------------
    int did_async = 0;
    if (use_pre) {
        // bf16 weight rows are 576B = 36 x 16B chunks; LDS rows padded to 584B (pad skipped).
#if HAVE_ASYNC_LDS
        for (int ci = tid; ci < J_ * 36; ci += 256) {          // 2304/256 = 9 iters, no tail
            int j  = ci / 36;
            int cc = ci - j * 36;
            g_v4i* gp = (g_v4i*)(wbf + ((g * J_ + j) * KDIM + cc * 8));
            l_v4i* lp = (l_v4i*)((char*)sA + j * (LDK * 2) + cc * 16);
            __builtin_amdgcn_global_load_async_to_lds_b128(gp, lp, 0, 0);
        }
        did_async = 1;
#else
        for (int ci = tid; ci < J_ * 72; ci += 256) {          // 8B chunks
            int j  = ci / 72;
            int cc = ci - j * 72;
            *(uint2*)((char*)sA + j * (LDK * 2) + cc * 8) =
                *(const uint2*)(wbf + ((g * J_ + j) * KDIM + cc * 4));
        }
#endif
    } else {
        const float* wg = wgt + (long)g * J_ * KDIM;
        for (int idx = tid; idx < J_ * KDIM; idx += 256) {
            int j = idx / KDIM;
            int k = idx - j * KDIM;
            sA[j * LDK + k] = f2bf(wg[j * KDIM + k]);
        }
    }

    // ---------------- Phase 1b: im2col gather (channel-permuted, zero-padded) ----------------
    // Block stride 256 is a multiple of NT=64 -> each thread's pixel is loop-invariant.
    {
        const int nl = tid & (NT - 1);
        const int n  = n0 + nl;
        const int w  = n % W_;
        const int h  = (n / W_) % H_;
        const int b  = n / (H_ * W_);
        const float* xb = x + (long)b * C_IN * H_ * W_;
        const int y0 = h - 1, x0 = w - 1;
        unsigned short* sBr = sB + nl * LDK;
        for (int k = tid >> 6; k < KDIM; k += 4) {             // 72 iters
            int c  = k / 9;
            int r  = k - c * 9;
            int kh = r / 3;
            int kw = r - kh * 3;
            int ch = arr[g * C_PER + c];
            int y  = y0 + kh;
            int xx = x0 + kw;
            float v = 0.0f;
            if ((unsigned)y < (unsigned)H_ && (unsigned)xx < (unsigned)W_)
                v = xb[(ch * H_ + y) * W_ + xx];
            sBr[k] = f2bf(v);
        }
    }

    if (did_async) wait_async0();
    __syncthreads();

    // ---------------- Phase 2: WMMA GEMM  (per wave: 16 M x 32 N tile) ----------------
    const int m_tile = wid & 3;      // 4 M tiles of 16  -> covers J=64
    const int n_half = wid >> 2;     // 2 N halves of 32 -> covers NT=64
    const int l15    = lane & 15;
    const int lhi    = lane >> 4;    // selects K half/quarter per ISA A/B layouts

    const char* sAc = (const char*)sA;
    const char* sBc = (const char*)sB;

    f32x8 acc0 = {0.f, 0.f, 0.f, 0.f, 0.f, 0.f, 0.f, 0.f};
    f32x8 acc1 = {0.f, 0.f, 0.f, 0.f, 0.f, 0.f, 0.f, 0.f};

#pragma unroll
    for (int ks = 0; ks < KSTEPS; ++ks) {
        // A frag: 16x32 bf16. Lane l: row M=l%16; v0..3 -> K=(l/16)*8+0..7,
        // v4..7 -> 16+(l/16)*8+0..7  => two 16B chunks at +0 and +32 bytes.
        FragU a;
        {
            const unsigned int* p = (const unsigned int*)
                (sAc + ((m_tile * 16 + l15) * (LDK * 2) + ks * 64 + lhi * 16));
            a.u[0] = p[0]; a.u[1] = p[1]; a.u[2] = p[2];  a.u[3] = p[3];
            a.u[4] = p[8]; a.u[5] = p[9]; a.u[6] = p[10]; a.u[7] = p[11];
        }
        // B frags: 32x16 bf16. Lane l: col N=l%16; K=(l/16)*8+0..7, 16+(l/16)*8+0..7 => two 16B chunks at +0 and +32 bytes.
        FragU b0, b1;
        {
            const unsigned int* p = (const unsigned int*)
                (sBc + (((n_half * 2 + 0) * 16 + l15) * (LDK * 2) + ks * 64 + lhi * 16));
#pragma unroll
            for (int i = 0; i < 8; ++i) b0.u[i] = p[i + (i >> 2) * 4];
        }
        {
            const unsigned int* p = (const unsigned int*)
                (sBc + (((n_half * 2 + 1) * 16 + l15) * (LDK * 2) + ks * 64 + lhi * 16));
#pragma unroll
            for (int i = 0; i < 8; ++i) b1.u[i] = p[i + (i >> 2) * 4];
        }
        acc0 = __builtin_amdgcn_wmma_f32_16x16x32_bf16(false, a.v, false, b0.v,
                                                       (short)0, acc0, false, false);
        acc1 = __builtin_amdgcn_wmma_f32_16x16x32_bf16(false, a.v, false, b1.v,
                                                       (short)0, acc1, false, false);
    }

    // ---------------- Epilogue: bias + store ----------------
    // C layout: VGPR i -> M = (lane/16)*8 + i, N = lane%16.
    float bs[8];
    {
        const float* bp = bias + g * J_ + m_tile * 16 + lhi * 8;
#pragma unroll
        for (int i = 0; i < 8; ++i) bs[i] = bp[i];
    }
#pragma unroll
    for (int t = 0; t < 2; ++t) {
        const f32x8 acc = t ? acc1 : acc0;
        int n = n0 + (n_half * 2 + t) * 16 + l15;
        int w = n % W_;
        int h = (n / W_) % H_;
        int b = n / (H_ * W_);
#pragma unroll
        for (int i = 0; i < 8; ++i) {
            int jl = m_tile * 16 + lhi * 8 + i;                 // 0..63 within group
            out[(((long)b * (G_ * J_) + g * J_ + jl) * H_ + h) * W_ + w] = acc[i] + bs[i];
        }
    }
}

extern "C" void kernel_launch(void* const* d_in, const int* in_sizes, int n_in,
                              void* d_out, int out_size, void* d_ws, size_t ws_size,
                              hipStream_t stream) {
    const float* x    = (const float*)d_in[0];
    const float* wgt  = (const float*)d_in[1];
    const float* bias = (const float*)d_in[2];
    const int*   arr  = (const int*)d_in[3];
    float*       out  = (float*)d_out;

    const int    n_w       = G_ * J_ * KDIM;                 // 147456 weights
    const size_t wbf_bytes = (size_t)n_w * sizeof(unsigned short);
    const int    use_pre   = (ws_size >= wbf_bytes) ? 1 : 0;
    unsigned short* wbf    = (unsigned short*)d_ws;

    if (use_pre) {
        convert_weights_bf16<<<dim3((n_w + 255) / 256), dim3(256), 0, stream>>>(wgt, wbf, n_w);
    }

    dim3 grid(NPIX / NT, G_);            // (784, 8)
    dim3 block(256);                     // 8 wave32
    size_t shmem = (size_t)(J_ + NT) * LDK * sizeof(unsigned short);   // 74,752 B
    grouped_conv_wmma_bf16<<<grid, block, shmem, stream>>>(x, wgt, bias, arr, wbf, use_pre, out);
}
